// CellFateNetTimeReversal_83854941487284
// MI455X (gfx1250) — matrix operand order repaired and checked by reading, not compiled, not run on hardware
//
#include <hip/hip_runtime.h>

// ---------------------------------------------------------------------------
// CDNA5 (gfx1250) implementation of CellFateNetTimeReversal.
// All matmuls run on v_wmma_f32_16x16x32_f16 (wave32, 16-row tiles).
// ---------------------------------------------------------------------------

typedef __attribute__((ext_vector_type(16))) _Float16 v16h;
typedef __attribute__((ext_vector_type(8)))  _Float16 v8h;
typedef __attribute__((ext_vector_type(8)))  float    v8f;

#define WMMA16x16x32(a, b, c) \
  __builtin_amdgcn_wmma_f32_16x16x32_f16(false, (a), false, (b), (short)0, (c), false, false)

// Wave-internal LDS fence: DS ops from one wave complete in order; this waits
// for outstanding DS traffic and stops the compiler reordering across it.
__device__ __forceinline__ void wave_lds_fence() {
  asm volatile("s_wait_dscnt 0" ::: "memory");
}

// A-fragment (16x32 f16, one K=32 chunk) from a contiguous 32-half row.
// ISA layout: lane l holds row (l&15), halfs K in [8*(l/16),+8) and [16+8*(l/16),+8).
__device__ __forceinline__ v16h frag_from_f16row(const _Float16* row, int hi8) {
  v8h lo = *(const v8h*)(row + hi8);
  v8h hg = *(const v8h*)(row + 16 + hi8);
  v16h a;
#pragma unroll
  for (int i = 0; i < 8; ++i) { a[i] = lo[i]; a[i + 8] = hg[i]; }
  return a;
}

// Same, but converting from an f32 row with a per-row scale (used for seg-mean).
__device__ __forceinline__ v16h frag_from_f32row(const float* row, int hi8, float s) {
  v16h a;
#pragma unroll
  for (int i = 0; i < 8; ++i) {
    a[i]     = (_Float16)(row[hi8 + i] * s);
    a[i + 8] = (_Float16)(row[16 + hi8 + i] * s);
  }
  return a;
}

// One linear layer: D[16 x NNC*16] = A[16 x NKC*32] * W + bias.
// Weight tiles are pre-packed per-lane (512 halfs/tile, tile index nc*NKC+kc).
template <int NKC, int NNC>
__device__ __forceinline__ void mlp_layer(const v16h* a, const _Float16* __restrict__ w,
                                          const float* __restrict__ bias, int lane, v8f* d) {
#pragma unroll
  for (int nc = 0; nc < NNC; ++nc) {
    float bv = bias[nc * 16 + (lane & 15)];
    v8f c;
#pragma unroll
    for (int r = 0; r < 8; ++r) c[r] = bv;
#pragma unroll
    for (int kc = 0; kc < NKC; ++kc) {
      v16h b = *(const v16h*)(w + (nc * NKC + kc) * 512 + lane * 16);
      c = WMMA16x16x32(a[kc], b, c);
    }
    d[nc] = c;
  }
}

// ReLU + store a 16x64 f32 D-block to LDS as f16 rows (row stride 64 halfs).
// D layout: lane l holds col n=l&15, rows m = r + 8*(l>>4).
__device__ __forceinline__ void relu_store_lds(_Float16* lds, const v8f* d, int lane) {
  int n = lane & 15, mb = (lane >> 4) * 8;
#pragma unroll
  for (int nc = 0; nc < 4; ++nc)
#pragma unroll
    for (int r = 0; r < 8; ++r)
      lds[(mb + r) * 64 + nc * 16 + n] = (_Float16)fmaxf(d[nc][r], 0.0f);
}

// Load both K=32 A-chunks of a 16x64 activation block from LDS.
__device__ __forceinline__ void lds_frags(const _Float16* lds, int lane, v16h* a) {
  int m = lane & 15, hi8 = (lane >> 4) * 8;
  const _Float16* row = lds + m * 64;
  a[0] = frag_from_f16row(row, hi8);
  a[1] = frag_from_f16row(row + 32, hi8);
}

// Write a 16x32 f32 D-pair as f16 rows to global (coalesced via LDS repack).
__device__ __forceinline__ void store_rows32(_Float16* lds, _Float16* out, int lane,
                                             const v8f& d0, const v8f& d1) {
  int n = lane & 15, mb = (lane >> 4) * 8;
#pragma unroll
  for (int r = 0; r < 8; ++r) {
    lds[(mb + r) * 32 + n]      = (_Float16)d0[r];
    lds[(mb + r) * 32 + 16 + n] = (_Float16)d1[r];
  }
  wave_lds_fence();
  const v8h* s = (const v8h*)lds;
  v8h* dp = (v8h*)out;
  dp[lane]      = s[lane];
  dp[lane + 32] = s[lane + 32];
}

// ---------------------------------------------------------------------------
// Kernels
// ---------------------------------------------------------------------------

// Pack one f32 weight matrix (K x Nout, row-major) into WMMA B-fragment tiles
// (f16) + padded f32 bias. B layout: lane l holds col n=l&15; half h -> K =
// kc*32 + 8*(l/16) + h for h<8, kc*32 + 16 + 8*(l/16) + (h-8) for h>=8.
__global__ void prep_layer_kernel(const float* __restrict__ W, const float* __restrict__ B,
                                  int K, int Nout, int Npad,
                                  _Float16* __restrict__ dW, float* __restrict__ dB) {
  int numKc = K >> 5;
  int t = blockIdx.x;                 // grid = numKc * (Npad/16)
  int kc = t % numKc, nc = t / numKc;
  int tid = threadIdx.x;              // 512 threads = one half each
  int lane = tid >> 4, h = tid & 15;
  int n = nc * 16 + (lane & 15);
  int k = kc * 32 + ((h < 8) ? h : h + 8) + ((lane >= 16) ? 8 : 0);
  float v = (k < K && n < Nout) ? W[k * Nout + n] : 0.0f;
  dW[(nc * numKc + kc) * 512 + lane * 16 + h] = (_Float16)v;
  if (t == 0 && tid < Npad) dB[tid] = (tid < Nout) ? B[tid] : 0.0f;
}

__global__ void zero2_kernel(float* a, int na, float* b, int nb) {
  int i = blockIdx.x * blockDim.x + threadIdx.x;
  int stride = gridDim.x * blockDim.x;
  for (int j = i; j < na; j += stride) a[j] = 0.0f;
  for (int j = i; j < nb; j += stride) b[j] = 0.0f;
}

// Encoder: x(f32,[rows,32]) -> 32->64->64->32 -> f16 rows.
__global__ void __launch_bounds__(256) encode_kernel(
    const float* __restrict__ x, long long nRows,
    const _Float16* __restrict__ w1, const float* __restrict__ b1,
    const _Float16* __restrict__ w2, const float* __restrict__ b2,
    const _Float16* __restrict__ w3, const float* __restrict__ b3,
    _Float16* __restrict__ outF) {
  __shared__ _Float16 ldsAll[8][1024];
  int lane = threadIdx.x & 31, wave = threadIdx.x >> 5;
  long long base = ((long long)blockIdx.x * 8 + wave) * 16;
  if (base >= nRows) return;
  _Float16* lds = ldsAll[wave];
  int m16 = lane & 15, hi8 = (lane >> 4) * 8;
  v16h a[2];
  a[0] = frag_from_f32row(x + (base + m16) * 32, hi8, 1.0f);
  v8f d[4];
  mlp_layer<1, 4>(a, w1, b1, lane, d);
  relu_store_lds(lds, d, lane);
  wave_lds_fence();
  lds_frags(lds, lane, a);
  wave_lds_fence();
  mlp_layer<2, 4>(a, w2, b2, lane, d);
  relu_store_lds(lds, d, lane);
  wave_lds_fence();
  lds_frags(lds, lane, a);
  wave_lds_fence();
  mlp_layer<2, 2>(a, w3, b3, lane, d);
  store_rows32(lds, outF + base * 32, lane, d[0], d[1]);
}

// Edge MLP: concat(feat0[idx0[e]], feat1[idx1[e]]) -> 64->64->64->32,
// then f32 atomic scatter-add into agg[scat[e]] plus edge count.
__global__ void __launch_bounds__(256) edge_kernel(
    const _Float16* __restrict__ feat0, const int* __restrict__ idx0,
    const _Float16* __restrict__ feat1, const int* __restrict__ idx1,
    const int* __restrict__ scat, long long nE,
    const _Float16* __restrict__ w1, const float* __restrict__ b1,
    const _Float16* __restrict__ w2, const float* __restrict__ b2,
    const _Float16* __restrict__ w3, const float* __restrict__ b3,
    float* __restrict__ agg, float* __restrict__ cnt) {
  __shared__ _Float16 ldsAll[8][1024];
  int lane = threadIdx.x & 31, wave = threadIdx.x >> 5;
  long long base = ((long long)blockIdx.x * 8 + wave) * 16;
  if (base >= nE) return;
  _Float16* lds = ldsAll[wave];
  int m16 = lane & 15, hi8 = (lane >> 4) * 8;
  long long e = base + m16;
  int i0 = idx0[e], i1 = idx1[e];
  v16h a[2];
  a[0] = frag_from_f16row(feat0 + (long long)i0 * 32, hi8);
  a[1] = frag_from_f16row(feat1 + (long long)i1 * 32, hi8);
  v8f d[4];
  mlp_layer<2, 4>(a, w1, b1, lane, d);
  relu_store_lds(lds, d, lane);
  wave_lds_fence();
  lds_frags(lds, lane, a);
  wave_lds_fence();
  mlp_layer<2, 4>(a, w2, b2, lane, d);
  relu_store_lds(lds, d, lane);
  wave_lds_fence();
  lds_frags(lds, lane, a);
  mlp_layer<2, 2>(a, w3, b3, lane, d);
  int n = m16, mb = hi8;
#pragma unroll
  for (int r = 0; r < 8; ++r) {
    int dn = scat[base + mb + r];
    atomicAdd(agg + (long long)dn * 32 + n,      d[0][r]);
    atomicAdd(agg + (long long)dn * 32 + 16 + n, d[1][r]);
  }
  if (lane < 16) atomicAdd(cnt + scat[base + lane], 1.0f);
}

// Node update: concat(featN[node], agg[node]/max(cnt,1)) -> 64->64->64->32.
__global__ void __launch_bounds__(256) node_kernel(
    const _Float16* __restrict__ featN,
    const float* __restrict__ agg, const float* __restrict__ cnt, long long nN,
    const _Float16* __restrict__ w1, const float* __restrict__ b1,
    const _Float16* __restrict__ w2, const float* __restrict__ b2,
    const _Float16* __restrict__ w3, const float* __restrict__ b3,
    _Float16* __restrict__ outF) {
  __shared__ _Float16 ldsAll[8][1024];
  int lane = threadIdx.x & 31, wave = threadIdx.x >> 5;
  long long base = ((long long)blockIdx.x * 8 + wave) * 16;
  if (base >= nN) return;
  _Float16* lds = ldsAll[wave];
  int m16 = lane & 15, hi8 = (lane >> 4) * 8;
  long long node = base + m16;
  float s = 1.0f / fmaxf(cnt[node], 1.0f);
  v16h a[2];
  a[0] = frag_from_f16row(featN + node * 32, hi8);
  a[1] = frag_from_f32row(agg + node * 32, hi8, s);
  v8f d[4];
  mlp_layer<2, 4>(a, w1, b1, lane, d);
  relu_store_lds(lds, d, lane);
  wave_lds_fence();
  lds_frags(lds, lane, a);
  wave_lds_fence();
  mlp_layer<2, 4>(a, w2, b2, lane, d);
  relu_store_lds(lds, d, lane);
  wave_lds_fence();
  lds_frags(lds, lane, a);
  mlp_layer<2, 2>(a, w3, b3, lane, d);
  store_rows32(lds, outF + base * 32, lane, d[0], d[1]);
}

// Decoder path: feat -> 32->64->64->16(pad of 3), returns one f32 D tile.
__device__ __forceinline__ v8f out_path(
    const _Float16* __restrict__ feat, long long base, int lane, _Float16* lds,
    const _Float16* w1, const float* b1, const _Float16* w2, const float* b2,
    const _Float16* w3, const float* b3) {
  int m16 = lane & 15, hi8 = (lane >> 4) * 8;
  v16h a[2];
  a[0] = frag_from_f16row(feat + (base + m16) * 32, hi8);
  v8f d[4];
  mlp_layer<1, 4>(a, w1, b1, lane, d);
  wave_lds_fence();
  relu_store_lds(lds, d, lane);
  wave_lds_fence();
  lds_frags(lds, lane, a);
  wave_lds_fence();
  mlp_layer<2, 4>(a, w2, b2, lane, d);
  relu_store_lds(lds, d, lane);
  wave_lds_fence();
  lds_frags(lds, lane, a);
  wave_lds_fence();
  v8f dout[1];
  mlp_layer<2, 1>(a, w3, b3, lane, dout);
  return dout[0];
}

__global__ void __launch_bounds__(256) out_kernel(
    const _Float16* __restrict__ hf, const _Float16* __restrict__ hr, long long nN,
    const _Float16* __restrict__ w1, const float* __restrict__ b1,
    const _Float16* __restrict__ w2, const float* __restrict__ b2,
    const _Float16* __restrict__ w3, const float* __restrict__ b3,
    float* __restrict__ out) {
  __shared__ _Float16 ldsAll[8][1024];
  int lane = threadIdx.x & 31, wave = threadIdx.x >> 5;
  long long base = ((long long)blockIdx.x * 8 + wave) * 16;
  if (base >= nN) return;
  _Float16* lds = ldsAll[wave];
  v8f da = out_path(hf, base, lane, lds, w1, b1, w2, b2, w3, b3);
  wave_lds_fence();
  v8f db = out_path(hr, base, lane, lds, w1, b1, w2, b2, w3, b3);
  int n = lane & 15, mb = (lane >> 4) * 8;
  if (n < 3) {
#pragma unroll
    for (int r = 0; r < 8; ++r)
      out[(base + mb + r) * 3 + n] = da[r] + db[r];
  }
}

// ---------------------------------------------------------------------------
// Host driver
// ---------------------------------------------------------------------------
#define WL(m, l) dW[(m) * 3 + (l)], dB[(m) * 3 + (l)]

extern "C" void kernel_launch(void* const* d_in, const int* in_sizes, int n_in,
                              void* d_out, int out_size, void* d_ws, size_t ws_size,
                              hipStream_t stream) {
  (void)out_size;
  constexpr long long T = 3, N = 50000, E = 800000, ET = 200000;

  const float* x    = (const float*)d_in[0];
  const int*   esrc = (const int*)d_in[1];
  const int*   edst = (const int*)d_in[2];
  const int*   tsrc = (const int*)d_in[3];
  const int*   tdst = (const int*)d_in[4];

  // Canonical MLP order: 0=enc 1=inter 2=node 3=time 4=time_node 5=time_rev
  //                      6=time_rev_node 7=out
  static const int dimsK[8][3] = {
      {32, 64, 64}, {64, 64, 64}, {64, 64, 64}, {64, 64, 64},
      {64, 64, 64}, {64, 64, 64}, {64, 64, 64}, {32, 64, 64}};
  static const int dimsN[8][3] = {
      {64, 64, 32}, {64, 64, 32}, {64, 64, 32}, {64, 64, 32},
      {64, 64, 32}, {64, 64, 32}, {64, 64, 32}, {64, 64, 3}};

  const float* Wsrc[24];
  const float* Bsrc[24];
  if (n_in >= 53) {
    // Separate leaf tensors. Detect param-dict flattening order:
    // in_sizes[23]==2048 -> alphabetical (out.W0), ==4096 -> insertion (time.W0).
    int mb[8];
    if (in_sizes[23] == 2048) { // enc,inter,node,out,time,time_node,time_rev,time_rev_node
      mb[0] = 5; mb[1] = 11; mb[2] = 17; mb[7] = 23;
      mb[3] = 29; mb[4] = 35; mb[5] = 41; mb[6] = 47;
    } else {                    // insertion order, out last
      mb[0] = 5; mb[1] = 11; mb[2] = 17; mb[3] = 23;
      mb[4] = 29; mb[5] = 35; mb[6] = 41; mb[7] = 47;
    }
    for (int m = 0; m < 8; ++m)
      for (int l = 0; l < 3; ++l) {
        Wsrc[m * 3 + l] = (const float*)d_in[mb[m] + 2 * l];
        Bsrc[m * 3 + l] = (const float*)d_in[mb[m] + 2 * l + 1];
      }
  } else {
    // params packed as one blob, assume JAX alphabetical leaf order.
    const float* blob = (const float*)d_in[5];
    static const int alphaIds[8] = {0, 1, 2, 7, 3, 4, 5, 6};
    size_t o = 0;
    for (int a = 0; a < 8; ++a) {
      int m = alphaIds[a];
      for (int l = 0; l < 3; ++l) {
        int K = dimsK[m][l], Nn = dimsN[m][l];
        Wsrc[m * 3 + l] = blob + o; o += (size_t)K * Nn;
        Bsrc[m * 3 + l] = blob + o; o += (size_t)Nn;
      }
    }
  }

  // Workspace carve-up.
  char* ws = (char*)d_ws;
  size_t off = 0;
  auto alloc = [&](size_t bytes) -> size_t {
    off = (off + 255) & ~(size_t)255;
    size_t o = off; off += bytes; return o;
  };
  _Float16* dW[24];
  float*    dB[24];
  int nTilesL[24], Kl[24], Nl[24], Npl[24];
  for (int i = 0; i < 24; ++i) {
    int m = i / 3, l = i % 3;
    int K = dimsK[m][l], Nn = dimsN[m][l];
    int Np = (Nn + 15) & ~15;
    int nt = (K / 32) * (Np / 16);
    Kl[i] = K; Nl[i] = Nn; Npl[i] = Np; nTilesL[i] = nt;
    dW[i] = (_Float16*)(ws + alloc((size_t)nt * 512 * 2));
    dB[i] = (float*)(ws + alloc((size_t)Np * 4));
  }
  _Float16* encF = (_Float16*)(ws + alloc((size_t)T * N * 32 * 2));
  _Float16* hF   = (_Float16*)(ws + alloc((size_t)T * N * 32 * 2));
  _Float16* hfA  = (_Float16*)(ws + alloc((size_t)N * 32 * 2));
  _Float16* hfB  = (_Float16*)(ws + alloc((size_t)N * 32 * 2));
  _Float16* hrA  = (_Float16*)(ws + alloc((size_t)N * 32 * 2));
  _Float16* hrB  = (_Float16*)(ws + alloc((size_t)N * 32 * 2));
  float*    agg  = (float*)(ws + alloc((size_t)N * 32 * 4));
  float*    cnt  = (float*)(ws + alloc((size_t)N * 4));
  if (off > ws_size) return;  // workspace too small: bail out safely

  // 1) Pack weights into WMMA fragment layout.
  for (int i = 0; i < 24; ++i)
    prep_layer_kernel<<<nTilesL[i], 512, 0, stream>>>(Wsrc[i], Bsrc[i], Kl[i], Nl[i], Npl[i],
                                                      dW[i], dB[i]);

  dim3 blk(256);
  auto blocksFor = [](long long items) -> int {
    long long tiles = (items + 15) / 16;
    return (int)((tiles + 7) / 8);
  };

  // 2) Encoder over all frames.
  encode_kernel<<<blocksFor(T * N), blk, 0, stream>>>(x, T * N, WL(0, 0), WL(0, 1), WL(0, 2), encF);

  // 3) Intra-frame interaction network.
  for (int t = 0; t < (int)T; ++t) {
    zero2_kernel<<<1024, 256, 0, stream>>>(agg, (int)(N * 32), cnt, (int)N);
    edge_kernel<<<blocksFor(E), blk, 0, stream>>>(
        encF + (long long)t * N * 32, esrc + (long long)t * E,
        encF + (long long)t * N * 32, edst + (long long)t * E,
        edst + (long long)t * E, E,
        WL(1, 0), WL(1, 1), WL(1, 2), agg, cnt);
    node_kernel<<<blocksFor(N), blk, 0, stream>>>(
        encF + (long long)t * N * 32, agg, cnt, N,
        WL(2, 0), WL(2, 1), WL(2, 2), hF + (long long)t * N * 32);
  }

  // 4) Forward-time propagation.
  const _Float16* hfPrev = hF;  // h[0]
  _Float16* hfOut[2] = {hfA, hfB};
  for (int t = 1; t < (int)T; ++t) {
    zero2_kernel<<<1024, 256, 0, stream>>>(agg, (int)(N * 32), cnt, (int)N);
    edge_kernel<<<blocksFor(ET), blk, 0, stream>>>(
        hfPrev, tsrc + (long long)(t - 1) * ET,
        hF + (long long)t * N * 32, tdst + (long long)(t - 1) * ET,
        tdst + (long long)(t - 1) * ET, ET,
        WL(3, 0), WL(3, 1), WL(3, 2), agg, cnt);
    node_kernel<<<blocksFor(N), blk, 0, stream>>>(
        hF + (long long)t * N * 32, agg, cnt, N,
        WL(4, 0), WL(4, 1), WL(4, 2), hfOut[t - 1]);
    hfPrev = hfOut[t - 1];
  }

  // 5) Time-reversal propagation.
  const _Float16* hrPrev = hF + (long long)(T - 1) * N * 32;  // h[T-1]
  _Float16* hrOut[2] = {hrA, hrB};
  int k = 0;
  for (int t = (int)T - 2; t >= 0; --t, ++k) {
    zero2_kernel<<<1024, 256, 0, stream>>>(agg, (int)(N * 32), cnt, (int)N);
    edge_kernel<<<blocksFor(ET), blk, 0, stream>>>(
        hrPrev, tdst + (long long)t * ET,                       // hr[d]
        hF + (long long)t * N * 32, tsrc + (long long)t * ET,   // h[t][s]
        tsrc + (long long)t * ET, ET,                           // seg over s
        WL(5, 0), WL(5, 1), WL(5, 2), agg, cnt);
    node_kernel<<<blocksFor(N), blk, 0, stream>>>(
        hF + (long long)t * N * 32, agg, cnt, N,
        WL(6, 0), WL(6, 1), WL(6, 2), hrOut[k]);
    hrPrev = hrOut[k];
  }

  // 6) Decode: out(hf) + out(hr) -> [N,3] f32.
  out_kernel<<<blocksFor(N), blk, 0, stream>>>(hfPrev, hrPrev, N,
                                               WL(7, 0), WL(7, 1), WL(7, 2), (float*)d_out);
}
